// BaseMLPExperts_60576218743041
// MI455X (gfx1250) — matrix operand order repaired and checked by reading, not compiled, not run on hardware
//
#include <hip/hip_runtime.h>
#include <math.h>

// ---------------------------------------------------------------------------
// Fused per-expert MLP:  out[t,e,:] = gelu(x[t,e,:] @ wi[e]) @ wo[e]
// E=64, T=2048, H=512, I=1024, all f32 in memory; f16 WMMA with f32 accum.
// ---------------------------------------------------------------------------

typedef _Float16 f16;
typedef __attribute__((ext_vector_type(16))) _Float16 v16h;
typedef __attribute__((ext_vector_type(8)))  float    v8f;
typedef unsigned int u32;
typedef __attribute__((ext_vector_type(4))) u32 u32x4;
typedef __attribute__((ext_vector_type(2))) u32 u32x2;

#define E_ 64
#define T_ 2048
#define H_ 512
#define I_ 1024

#define TM   32      // token tile per workgroup (M)
#define IK   128     // i-block: 8 waves x one 16-wide i-tile each
#define XPAD 8       // LDS padding (halves) to spread banks
#define HPAD 8

union Frag  { v16h v; u32x4 u[2]; };
union Pack4 { f16 h[4]; u32x2 d; };

// ---------------------------------------------------------------------------
// Tiled transpose + f32 -> f16 convert:  in [B][R][C] f32  ->  out [B][C][R] f16
// ---------------------------------------------------------------------------
__global__ __launch_bounds__(256)
void transpose_cvt_f16(const float* __restrict__ in, f16* __restrict__ out,
                       int R, int C)
{
    __shared__ float tile[32][33];
    const int bx = blockIdx.x, by = blockIdx.y, z = blockIdx.z;
    const int tx = threadIdx.x & 31;
    const int ty = threadIdx.x >> 5;            // 0..7
    const float* ip = in  + (size_t)z * R * C;
    f16*         op = out + (size_t)z * R * C;

#pragma unroll
    for (int i = 0; i < 32; i += 8)
        tile[ty + i][tx] = ip[(size_t)(by * 32 + ty + i) * C + bx * 32 + tx];
    __syncthreads();
#pragma unroll
    for (int i = 0; i < 32; i += 8)
        op[(size_t)(bx * 32 + ty + i) * R + by * 32 + tx] = (f16)tile[tx][ty + i];
}

// ---------------------------------------------------------------------------
// Fused expert MLP. Grid: (T/TM, E). Block: 256 threads = 8 waves (wave32).
//   wiT : [E][I][H] f16   (B of GEMM1, N-major so K runs contiguous per lane)
//   woT : [E][H][I] f16   (B of GEMM2)
// Per i-block (128 wide): each wave computes a 32x16 slice of gelu(h) (two
// M-tiles sharing each B fragment -> B loaded once, 2 independent WMMA
// chains), then all waves consume hs for their 2x4 out tiles.
// ---------------------------------------------------------------------------
__global__ __launch_bounds__(256)
void moe_mlp_fused(const float* __restrict__ x,
                   const f16* __restrict__ wiT,
                   const f16* __restrict__ woT,
                   float* __restrict__ out)
{
    __shared__ __align__(16) f16 xs[TM][H_ + XPAD];   // x tile, f16
    __shared__ __align__(16) f16 hs[TM][IK + HPAD];   // gelu(h) block, f16

    const int e   = blockIdx.y;
    const int t0  = blockIdx.x * TM;
    const int tid = threadIdx.x;
    const int w   = tid >> 5;          // wave 0..7
    const int l   = tid & 31;
    const int l16 = l & 15;
    const int hi  = (l >> 4) & 1;      // lane half: 0 = lanes 0-15, 1 = 16-31

    // ---- stage x tile (32 x 512 f32) into LDS as f16 -----------------------
    {
        const int cx = (tid & 127) * 4;      // float4 column
        const int r0 = tid >> 7;             // 0..1
        for (int r = r0; r < TM; r += 2) {
            const float4 f =
                *(const float4*)&x[((size_t)(t0 + r) * E_ + e) * H_ + cx];
            Pack4 p;
            p.h[0] = (f16)f.x; p.h[1] = (f16)f.y;
            p.h[2] = (f16)f.z; p.h[3] = (f16)f.w;
            *(u32x2*)&xs[r][cx] = p.d;       // one 8-byte LDS store
        }
    }
    __syncthreads();

    // output accumulators: wave w owns out[mt*16.., w*64 + nt*16..], 2x4 tiles
    v8f oacc[2][4];
#pragma unroll
    for (int a = 0; a < 2; ++a)
#pragma unroll
        for (int b = 0; b < 4; ++b)
            oacc[a][b] = (v8f){};

    const int n0 = w * 64;             // GEMM2: wave's H column base

    for (int ib = 0; ib < I_ / IK; ++ib) {
        // ------- GEMM1: h[:, ib*128 + w*16 .. +16] = x @ wi ----------------
        // wave w owns i-tile w; computes both M-tiles; B loaded once per ks.
        v8f hacc[2];
        hacc[0] = (v8f){};
        hacc[1] = (v8f){};
        const int nidx = ib * IK + w * 16 + l16;       // global i (B column)
        const f16* bbase = &wiT[((size_t)e * I_ + nidx) * H_];
        const f16* a0base = &xs[l16][0];               // mt = 0 row
        const f16* a1base = &xs[16 + l16][0];          // mt = 1 row
#pragma unroll 4
        for (int ks = 0; ks < H_ / 32; ++ks) {
            Frag B, A0, A1;
            const f16* bp = bbase + ks * 32 + hi * 8;
            B.u[0] = *(const u32x4*)bp;
            B.u[1] = *(const u32x4*)(bp + 16);
            const int ac = ks * 32 + hi * 8;
            A0.u[0] = *(const u32x4*)(a0base + ac);
            A0.u[1] = *(const u32x4*)(a0base + ac + 16);
            A1.u[0] = *(const u32x4*)(a1base + ac);
            A1.u[1] = *(const u32x4*)(a1base + ac + 16);
            hacc[0] = __builtin_amdgcn_wmma_f32_16x16x32_f16(
                false, A0.v, false, B.v, (short)0, hacc[0], false, false);
            hacc[1] = __builtin_amdgcn_wmma_f32_16x16x32_f16(
                false, A1.v, false, B.v, (short)0, hacc[1], false, false);
        }

        // exact GELU, convert to f16, scatter into hs (D layout: m = hi*8+r)
#pragma unroll
        for (int mt = 0; mt < 2; ++mt) {
#pragma unroll
            for (int r = 0; r < 8; ++r) {
                float v = hacc[mt][r];
                v = 0.5f * v * (1.0f + erff(v * 0.70710678118654752f));
                hs[mt * 16 + hi * 8 + r][w * 16 + l16] = (f16)v;
            }
        }
        __syncthreads();   // hs fully written before any wave reads it

        // ------- GEMM2: out += gelu(h) @ wo --------------------------------
#pragma unroll
        for (int kk = 0; kk < IK / 32; ++kk) {
            Frag A2[2], B2[4];
#pragma unroll
            for (int mt = 0; mt < 2; ++mt) {
                const f16* ap = &hs[mt * 16 + l16][kk * 32 + hi * 8];
                A2[mt].u[0] = *(const u32x4*)ap;
                A2[mt].u[1] = *(const u32x4*)(ap + 16);
            }
#pragma unroll
            for (int nt = 0; nt < 4; ++nt) {
                const f16* bp =
                    &woT[((size_t)e * H_ + n0 + nt * 16 + l16) * I_ +
                         ib * IK + kk * 32 + hi * 8];
                B2[nt].u[0] = *(const u32x4*)bp;
                B2[nt].u[1] = *(const u32x4*)(bp + 16);
            }
#pragma unroll
            for (int mt = 0; mt < 2; ++mt)
#pragma unroll
                for (int nt = 0; nt < 4; ++nt)
                    oacc[mt][nt] = __builtin_amdgcn_wmma_f32_16x16x32_f16(
                        false, A2[mt].v, false, B2[nt].v,
                        (short)0, oacc[mt][nt], false, false);
        }
        __syncthreads();   // all reads of hs done before next iter overwrites
    }

    // ---- write out tile: (m,n) lives in VGPR (m%8), lane (m/8)*16 + n ------
#pragma unroll
    for (int mt = 0; mt < 2; ++mt) {
        const int m = mt * 16 + hi * 8;
#pragma unroll
        for (int nt = 0; nt < 4; ++nt) {
            const int n = n0 + nt * 16 + l16;
#pragma unroll
            for (int r = 0; r < 8; ++r)
                out[((size_t)(t0 + m + r) * E_ + e) * H_ + n] = oacc[mt][nt][r];
        }
    }
}

// ---------------------------------------------------------------------------
extern "C" void kernel_launch(void* const* d_in, const int* in_sizes, int n_in,
                              void* d_out, int out_size, void* d_ws, size_t ws_size,
                              hipStream_t stream)
{
    (void)in_sizes; (void)n_in; (void)out_size; (void)ws_size;

    const float* x  = (const float*)d_in[0];   // [T, E, H] f32
    const float* wi = (const float*)d_in[1];   // [E, H, I] f32
    const float* wo = (const float*)d_in[2];   // [E, I, H] f32
    float* out = (float*)d_out;                // [T, E, H] f32

    // workspace: wiT [E][I][H] f16, woT [E][H][I] f16  (134 MB total)
    f16* wiT = (f16*)d_ws;
    f16* woT = wiT + (size_t)E_ * H_ * I_;

    // wi [E][H][I] -> wiT [E][I][H]   (R=H, C=I)
    transpose_cvt_f16<<<dim3(I_ / 32, H_ / 32, E_), 256, 0, stream>>>(wi, wiT, H_, I_);
    // wo [E][I][H] -> woT [E][H][I]   (R=I, C=H)
    transpose_cvt_f16<<<dim3(H_ / 32, I_ / 32, E_), 256, 0, stream>>>(wo, woT, I_, H_);

    moe_mlp_fused<<<dim3(T_ / TM, E_), 256, 0, stream>>>(x, wiT, woT, out);
}
